// AntiSymmetric_43654047596706
// MI455X (gfx1250) — matrix operand order repaired and checked by reading, not compiled, not run on hardware
//
#include <hip/hip_runtime.h>
#include <math.h>

// ---------------------------------------------------------------------------
// AntiSymmetricConv GNN for MI455X (gfx1250, wave32, WMMA).
//  - All dense [N,128]x[128,128] matmuls run through v_wmma_f32_16x16x32_f16
//    (f16 inputs, f32 accumulate), K=128 = 4 WMMA steps, 16 WMMAs/wave.
//  - Weight operands pre-transposed to column-major f16 so every WMMA
//    fragment is a contiguous 32B run in LDS (2x ds_load_b128 per fragment).
//  - LDS tiles padded to 272B row stride -> bank-conflict-free fragment reads.
//  - Edge aggregation = coalesced float4 gather + f32 atomics into an
//    L2-resident [N,128] accumulator (51MB working set << 192MB L2).
// ---------------------------------------------------------------------------

typedef __attribute__((ext_vector_type(16))) _Float16 v16h;
typedef __attribute__((ext_vector_type(8)))  float    v8f;

#define GAMMA 0.1f
#define EPS   0.1f
#define LDT   136   // padded LDS row stride in halves (272B = 16B-aligned, 4-bank rotate)

// ---------------- elementwise / graph-prep kernels -------------------------

__global__ __launch_bounds__(256) void zero_f32_kernel(float* __restrict__ p, int n) {
    int i = blockIdx.x * 256 + threadIdx.x;
    if (i < n) p[i] = 0.f;
}

__global__ __launch_bounds__(256) void degree_kernel(const int* __restrict__ dst,
                                                     float* __restrict__ deg, int nE) {
    int e = blockIdx.x * 256 + threadIdx.x;
    if (e < nE) atomicAdd(&deg[dst[e]], 1.0f);
}

__global__ __launch_bounds__(256) void dinv_kernel(const float* __restrict__ deg,
                                                   float* __restrict__ dinv, int n) {
    int i = blockIdx.x * 256 + threadIdx.x;
    if (i < n) dinv[i] = rsqrtf(deg[i] + 1.0f);
}

// Build the f16 weight operand TRANSPOSED: Bt[n][k] = B[k][n] where C = A @ B.
// mode 0: B[k][n] = S[k][n]                      (x @ theta)
// mode 1: B[k][n] = S[n][k]                      (h @ lw1^T)   -> Bt = straight copy
// mode 2: B[k][n] = S[n][k] - S[k][n] - g*(k==n) (x @ M^T, M = W - W^T - g*I)
__global__ __launch_bounds__(256) void prep_b_kernel(const float* __restrict__ S,
                                                     _Float16* __restrict__ Bt,
                                                     int mode, float gamma) {
    int i = blockIdx.x * 256 + threadIdx.x;   // 0..16383; Bt[i] with n=i>>7, k=i&127
    int n = i >> 7, k = i & 127;
    float v;
    if (mode == 0)      v = S[k * 128 + n];
    else if (mode == 1) v = S[n * 128 + k];
    else                v = S[n * 128 + k] - S[k * 128 + n] - ((k == n) ? gamma : 0.f);
    Bt[i] = (_Float16)v;
}

// agg[i][f] = xw[i][f] * dinv[i]^2   (self-loop term, initializes agg)
__global__ __launch_bounds__(256) void selfloop_kernel(const float* __restrict__ xw,
                                                       const float* __restrict__ dinv,
                                                       float* __restrict__ agg, int n) {
    int i = blockIdx.x * 256 + threadIdx.x;   // over n*32 float4s
    if (i >= n * 32) return;
    int node = i >> 5;
    float s = dinv[node]; s = s * s;
    float4 v = ((const float4*)xw)[i];
    v.x *= s; v.y *= s; v.z *= s; v.w *= s;
    ((float4*)agg)[i] = v;
}

// One wave32 per edge: lane l handles floats [4l,4l+4) of the 128-wide row.
__global__ __launch_bounds__(256) void edge_scatter_kernel(const float* __restrict__ xw,
                                                           const float* __restrict__ dinv,
                                                           const int* __restrict__ src,
                                                           const int* __restrict__ dst,
                                                           float* __restrict__ agg, int nE) {
    int w    = (blockIdx.x * 256 + threadIdx.x) >> 5;
    int lane = threadIdx.x & 31;
    if (w >= nE) return;
    int s = src[w], d = dst[w];
    float nrm = dinv[s] * dinv[d];
    float4 v = ((const float4*)xw)[(size_t)s * 32 + lane];
    float* p = agg + (size_t)d * 128 + (size_t)lane * 4;
    atomicAdd(p + 0, v.x * nrm);
    atomicAdd(p + 1, v.y * nrm);
    atomicAdd(p + 2, v.z * nrm);
    atomicAdd(p + 3, v.w * nrm);
}

// out = relu(x + EPS * tanh(xm + agg + bias[f]))
__global__ __launch_bounds__(256) void combine_kernel(const float* __restrict__ x,
                                                      const float* __restrict__ xm,
                                                      const float* __restrict__ agg,
                                                      const float* __restrict__ bias,
                                                      float* __restrict__ out, int total) {
    int i = blockIdx.x * 256 + threadIdx.x;
    if (i >= total) return;
    float z = xm[i] + agg[i] + bias[i & 127];
    float v = x[i] + EPS * tanhf(z);
    out[i] = fmaxf(v, 0.f);
}

// ---------------- WMMA GEMM: C[rows,128] = A[rows,128] @ B[128,128] --------
// B is supplied TRANSPOSED (Bt[n][k], f16). Block = 256 threads (8 waves);
// block computes a 64x128 output tile; wave w owns columns [16w,16w+16) and
// iterates 4 row-subtiles -> 16 WMMAs / wave. All fragment reads are
// contiguous 16B LDS loads with 272B lane stride (conflict-free).
__global__ __launch_bounds__(256) void gemm_wmma_n128(const float* __restrict__ A,
                                                      const _Float16* __restrict__ Bt,
                                                      const float* __restrict__ bias,
                                                      float* __restrict__ C,
                                                      int rows, int relu) {
    __shared__ _Float16 sB[128 * LDT];  // ~34 KB: Bt[n][k] padded
    __shared__ _Float16 sA[64 * LDT];   // ~17 KB: 64-row A tile (f32 -> f16), padded

    const int tid  = threadIdx.x;
    const int row0 = blockIdx.x * 64;

    // Stage Bt: 2048 uint4 (8 halves each) / 256 threads = 8 each.
    #pragma unroll
    for (int i = 0; i < 8; ++i) {
        int idx = tid + i * 256;          // uint4 index; 16 per 128-half row
        int r = idx >> 4, c = idx & 15;
        *(uint4*)&sB[r * LDT + c * 8] = ((const uint4*)Bt)[idx];
    }
    // Stage + convert A: 2048 float4 / 256 threads = 8 each.
    #pragma unroll
    for (int i = 0; i < 8; ++i) {
        int idx = tid + i * 256;          // float4 index in 64x128 tile
        int r = idx >> 5, c4 = idx & 31;  // 32 float4 per row
        int gr = row0 + r;
        float4 v = make_float4(0.f, 0.f, 0.f, 0.f);
        if (gr < rows) v = ((const float4*)A)[(size_t)gr * 32 + c4];
        _Float16* p = &sA[r * LDT + c4 * 4];
        p[0] = (_Float16)v.x; p[1] = (_Float16)v.y;
        p[2] = (_Float16)v.z; p[3] = (_Float16)v.w;
    }
    __syncthreads();

    const int lane = tid & 31;
    const int nIdx = lane & 15;   // column within 16-wide tile / A subtile row
    const int hi   = lane >> 4;   // lane half selects K sub-range (ISA 7.12.2)
    const int col0 = (tid >> 5) * 16;

    v8f acc[4];
    #pragma unroll
    for (int rt = 0; rt < 4; ++rt)
        acc[rt] = (v8f){0.f, 0.f, 0.f, 0.f, 0.f, 0.f, 0.f, 0.f};

    #pragma unroll
    for (int kk = 0; kk < 4; ++kk) {
        const int kb = kk * 32;
        // B fragment: lane half hi covers K = kb+16*hi .. +15 at fixed column
        // -> 16 contiguous halves (32B) in transposed LDS image.
        const _Float16* bp = &sB[(col0 + nIdx) * LDT + kb + hi * 16];
        v16h b;
        #pragma unroll
        for (int i = 0; i < 16; ++i) b[i] = bp[i];
        #pragma unroll
        for (int rt = 0; rt < 4; ++rt) {
            // A fragment: K = kb + hi*16 + i -> 16 contiguous halves.
            const _Float16* ap = &sA[(rt * 16 + nIdx) * LDT + kb + hi * 16];
            v16h a;
            #pragma unroll
            for (int i = 0; i < 16; ++i) a[i] = ap[i];
            acc[rt] = __builtin_amdgcn_wmma_f32_16x16x32_f16(
                false, a, false, b, (short)0, acc[rt], false, false);
        }
    }

    // C/D layout: VGPR j -> M = j + 8*hi, N = nIdx.
    #pragma unroll
    for (int rt = 0; rt < 4; ++rt) {
        #pragma unroll
        for (int j = 0; j < 8; ++j) {
            int r = row0 + rt * 16 + hi * 8 + j;
            if (r < rows) {
                float v = acc[rt][j];
                if (bias) v += bias[col0 + nIdx];
                if (relu) v = fmaxf(v, 0.f);
                C[(size_t)r * 128 + col0 + nIdx] = v;
            }
        }
    }
}

// ---------------- classifier + log_softmax (one wave32 per node) -----------
__global__ __launch_bounds__(128) void logits_lsm_kernel(const float* __restrict__ h,
                                                         const float* __restrict__ lw2,
                                                         const float* __restrict__ lb2,
                                                         float* __restrict__ out,
                                                         int nNodes, int nCls) {
    __shared__ float sh[4][128];
    const int wave = threadIdx.x >> 5, lane = threadIdx.x & 31;
    const int node = blockIdx.x * 4 + wave;
    const bool valid = node < nNodes;
    if (valid)
        ((float4*)sh[wave])[lane] = ((const float4*)h)[(size_t)node * 32 + lane];
    __syncthreads();
    if (!valid) return;

    // class c0 = lane (always < 40); class c1 = 32+lane (valid for lane < 8)
    float lg0 = 0.f, lg1 = -INFINITY;
    const float4* hv = (const float4*)sh[wave];
    {
        const float4* w = (const float4*)(lw2 + (size_t)lane * 128);
        float a = 0.f;
        #pragma unroll 8
        for (int k = 0; k < 32; ++k) {
            float4 x = hv[k], y = w[k];
            a += x.x * y.x + x.y * y.y + x.z * y.z + x.w * y.w;
        }
        lg0 = a + lb2[lane];
    }
    if (lane < 8 && (32 + lane) < nCls) {
        const float4* w = (const float4*)(lw2 + (size_t)(32 + lane) * 128);
        float a = 0.f;
        #pragma unroll 8
        for (int k = 0; k < 32; ++k) {
            float4 x = hv[k], y = w[k];
            a += x.x * y.x + x.y * y.y + x.z * y.z + x.w * y.w;
        }
        lg1 = a + lb2[32 + lane];
    }
    float m = fmaxf(lg0, lg1);
    #pragma unroll
    for (int o = 16; o > 0; o >>= 1) m = fmaxf(m, __shfl_xor(m, o, 32));
    float s = expf(lg0 - m) + ((lg1 == -INFINITY) ? 0.f : expf(lg1 - m));
    #pragma unroll
    for (int o = 16; o > 0; o >>= 1) s += __shfl_xor(s, o, 32);
    float lse = m + logf(s);
    out[(size_t)node * nCls + lane] = lg0 - lse;
    if (lane < 8 && (32 + lane) < nCls)
        out[(size_t)node * nCls + 32 + lane] = lg1 - lse;
}

// ---------------------------------------------------------------------------

extern "C" void kernel_launch(void* const* d_in, const int* in_sizes, int n_in,
                              void* d_out, int out_size, void* d_ws, size_t ws_size,
                              hipStream_t stream) {
    const float* x   = (const float*)d_in[0];
    const int*   ei  = (const int*)  d_in[1];
    const float* W1  = (const float*)d_in[2];
    const float* b1  = (const float*)d_in[3];
    const float* T1  = (const float*)d_in[4];
    const float* W2  = (const float*)d_in[5];
    const float* b2  = (const float*)d_in[6];
    const float* T2  = (const float*)d_in[7];
    const float* lw1 = (const float*)d_in[8];
    const float* lb1 = (const float*)d_in[9];
    const float* lw2 = (const float*)d_in[10];
    const float* lb2 = (const float*)d_in[11];

    const int N = in_sizes[0] / 128;
    const int E = in_sizes[1] / 2;
    const int C = in_sizes[10] / 128;
    const int* src = ei;
    const int* dst = ei + E;

    // workspace carve-up (256B aligned)
    char* w = (char*)d_ws;
    auto alloc = [&](size_t bytes) -> void* {
        void* p = (void*)w;
        w += (bytes + 255) & ~(size_t)255;
        return p;
    };
    float*    deg  = (float*)alloc((size_t)N * 4);
    float*    dinv = (float*)alloc((size_t)N * 4);
    _Float16* BT1  = (_Float16*)alloc(128 * 128 * 2);
    _Float16* BM1  = (_Float16*)alloc(128 * 128 * 2);
    _Float16* BL1  = (_Float16*)alloc(128 * 128 * 2);
    _Float16* BT2  = (_Float16*)alloc(128 * 128 * 2);
    _Float16* BM2  = (_Float16*)alloc(128 * 128 * 2);
    float* B0 = (float*)alloc((size_t)N * 128 * 4);
    float* B1 = (float*)alloc((size_t)N * 128 * 4);
    float* B2 = (float*)alloc((size_t)N * 128 * 4);
    float* B3 = (float*)alloc((size_t)N * 128 * 4);

    float* result = (float*)d_out;                  // [N, C] log-softmax
    float* x1     = (float*)d_out + (size_t)N * C;  // [N, 128] relu(conv2)

    const int gN   = (N + 255) / 256;
    const int gE   = (E + 255) / 256;
    const int gEl  = (N * 128 + 255) / 256;
    const int gV4  = (N * 32 + 255) / 256;
    const int gEdg = (E * 32 + 255) / 256;    // one wave per edge
    const int gGmm = (N + 63) / 64;

    // graph normalization (shared by both conv layers)
    zero_f32_kernel<<<gN, 256, 0, stream>>>(deg, N);
    degree_kernel<<<gE, 256, 0, stream>>>(dst, deg, E);
    dinv_kernel<<<gN, 256, 0, stream>>>(deg, dinv, N);

    // f16 weight operands (transposed layout)
    prep_b_kernel<<<64, 256, 0, stream>>>(T1,  BT1, 0, GAMMA);
    prep_b_kernel<<<64, 256, 0, stream>>>(W1,  BM1, 2, GAMMA);
    prep_b_kernel<<<64, 256, 0, stream>>>(lw1, BL1, 1, GAMMA);
    prep_b_kernel<<<64, 256, 0, stream>>>(T2,  BT2, 0, GAMMA);
    prep_b_kernel<<<64, 256, 0, stream>>>(W2,  BM2, 2, GAMMA);

    // ---- AntiSymmetricConv 1 ----
    gemm_wmma_n128<<<gGmm, 256, 0, stream>>>(x, BT1, nullptr, B0, N, 0); // xw
    gemm_wmma_n128<<<gGmm, 256, 0, stream>>>(x, BM1, nullptr, B1, N, 0); // x@M^T
    selfloop_kernel<<<gV4, 256, 0, stream>>>(B0, dinv, B2, N);
    edge_scatter_kernel<<<gEdg, 256, 0, stream>>>(B0, dinv, src, dst, B2, E);
    combine_kernel<<<gEl, 256, 0, stream>>>(x, B1, B2, b1, B3, N * 128); // h1 (relu)

    // ---- Linear 128->128 + relu ----
    gemm_wmma_n128<<<gGmm, 256, 0, stream>>>(B3, BL1, lb1, B0, N, 1);    // h2

    // ---- AntiSymmetricConv 2 ----
    gemm_wmma_n128<<<gGmm, 256, 0, stream>>>(B0, BT2, nullptr, B1, N, 0); // xw2
    gemm_wmma_n128<<<gGmm, 256, 0, stream>>>(B0, BM2, nullptr, B3, N, 0); // h2@M^T
    selfloop_kernel<<<gV4, 256, 0, stream>>>(B1, dinv, B2, N);
    edge_scatter_kernel<<<gEdg, 256, 0, stream>>>(B1, dinv, src, dst, B2, E);
    combine_kernel<<<gEl, 256, 0, stream>>>(B0, B3, B2, b2, x1, N * 128); // x1 (relu)

    // ---- classifier + log_softmax ----
    logits_lsm_kernel<<<(N + 3) / 4, 128, 0, stream>>>(x1, lw2, lb2, result, N, C);
}
